// OtLlamaAttention_81733227643254
// MI455X (gfx1250) — matrix operand order repaired and checked by reading, not compiled, not run on hardware
//
#include <hip/hip_runtime.h>
#include <hip/hip_bf16.h>
#include <stdint.h>

// ---------------- problem constants ----------------
#define SEQ   2048
#define DH    128
#define NH    32
#define NKV   8
#define OUT0_N (SEQ*NH*DH)            // 8,388,608 attn_output floats
static constexpr float SCALING_ = 0.08838834764831843f;   // 128^-0.5
static constexpr float QLUTMIN_ = 0.02004f;

// ---------------- vector types for CDNA5 builtins ----------------
typedef __attribute__((ext_vector_type(16))) __bf16 v16bf;
typedef __attribute__((ext_vector_type(8)))  float  v8f;
typedef __attribute__((ext_vector_type(4)))  unsigned int v4u;
typedef __attribute__((ext_vector_type(8)))  unsigned int v8u;
typedef __attribute__((ext_vector_type(8)))  int    v8i_t;
typedef __attribute__((ext_vector_type(4)))  int    v4i_t;

// ---------------- small helpers ----------------
// order-preserving float<->u32 encoding so min/max can use u32 atomics
__device__ inline unsigned fenc(float f){
  unsigned u = __float_as_uint(f);
  return (u & 0x80000000u) ? ~u : (u | 0x80000000u);
}
__device__ inline float fdec(unsigned e){
  unsigned u = (e & 0x80000000u) ? (e & 0x7FFFFFFFu) : ~e;
  return __uint_as_float(u);
}
__device__ inline unsigned hash_u(unsigned x){
  x ^= x >> 16; x *= 0x7feb352du; x ^= x >> 15; x *= 0x846ca68bu; x ^= x >> 16;
  return x;
}
__device__ inline float urand01(unsigned idx, unsigned salt){
  return (float)(hash_u(idx ^ salt) >> 8) * (1.0f/16777216.0f);
}

// quantization params resolved once per kernel: no per-element division
struct QParams { float lo, hi, scale, inv; };
__device__ inline QParams qload(const float* mm){
  QParams p; p.lo = mm[0]; p.hi = mm[1];
  p.scale = fmaxf((p.hi - p.lo) * (1.0f/255.0f), 1e-30f);
  p.inv = 1.0f / p.scale;               // single division per kernel
  return p;
}
// deterministic uniform fake-quant (256 levels, round-half-even like jnp.round)
__device__ inline float fq_u(float x, const QParams& p){
  float xc = fminf(fmaxf(x, p.lo), p.hi);
  return rintf((xc - p.lo) * p.inv) * p.scale + p.lo;
}
// weight-side quant: |q| below thresh (= lut_min * full_scale) -> 0
__device__ inline float fq_lut(float b, const QParams& p, float thresh){
  float q = fq_u(b, p);
  return (fabsf(q) < thresh) ? 0.0f : q;
}
__device__ inline unsigned short bfb(float x){ return __builtin_bit_cast(unsigned short, (__bf16)x); }
__device__ inline unsigned pk2(float a, float b){ return ((unsigned)bfb(b) << 16) | (unsigned)bfb(a); }

// fragment loads straight out of LDS, no 16-bit repacking:
// 8 consecutive u32 (16 consecutive bf16) -> v16bf
__device__ inline v16bf frag8(const unsigned* p){
  v8u u = *(const v8u*)p;                       // 2x ds_load_b128
  return __builtin_bit_cast(v16bf, u);
}
// two groups of 4 consecutive u32 -> v16bf
struct V4x2 { v4u a, b; };
__device__ inline v16bf frag44(const unsigned* pa, const unsigned* pb){
  V4x2 t; t.a = *(const v4u*)pa; t.b = *(const v4u*)pb;   // 2x ds_load_b128
  return __builtin_bit_cast(v16bf, t);
}

__device__ inline float wave_min(float v){ for(int o=16;o;o>>=1) v=fminf(v,__shfl_xor(v,o)); return v; }
__device__ inline float wave_max(float v){ for(int o=16;o;o>>=1) v=fmaxf(v,__shfl_xor(v,o)); return v; }
__device__ inline float wave_sum(float v){ for(int o=16;o;o>>=1) v+=__shfl_xor(v,o); return v; }

// flat shared-pointer -> LDS byte offset (LDS aperture keeps offset in addr[31:0])
__device__ inline unsigned lds_off(const void* p){
  return (unsigned)(unsigned long long)p;
}

// ---------------- Tensor Data Mover (TDM) ----------------
// Pack D# group0 (128b) + group1 (256b) per CDNA5 ISA.
// 2D tile of f32: tile_w x tile_h elements, row stride in elements.
// 1D load: tile_h = 0 (unused), tile_w = element count.
__device__ inline void tdm_desc(v4u& g0, v8i_t& g1, unsigned ldsByteOff,
                                const void* gp, unsigned tile_w, unsigned tile_h,
                                unsigned long long row_stride, unsigned tensor_w){
  unsigned long long ga = (unsigned long long)gp;
  g0.x = 1u;                                                // count=1, user mode
  g0.y = ldsByteOff;                                        // lds_addr [63:32]
  g0.z = (unsigned)ga;                                      // global_addr[31:0]
  g0.w = (unsigned)((ga >> 32) & 0x01FFFFFFull) | (2u<<30); // addr[56:32] | type=2
  const unsigned td1 = 1u << 20;                            // tensor_dim1 (large)
  g1[0] = (int)(2u << 16);                                  // data_size=2 (4 bytes)
  g1[1] = (int)((tensor_w & 0xFFFFu) << 16);                // tensor_dim0[15:0]
  g1[2] = (int)(((tensor_w >> 16) & 0xFFFFu) | ((td1 & 0xFFFFu) << 16));
  g1[3] = (int)(((td1 >> 16) & 0xFFFFu) | ((tile_w & 0xFFFFu) << 16)); // tile_dim0
  g1[4] = (int)(tile_h & 0xFFFFu);                          // tile_dim1 (dim2=0)
  g1[5] = (int)(row_stride & 0xFFFFFFFFull);                // tensor_dim0_stride lo
  g1[6] = (int)((row_stride >> 32) & 0xFFFFull);            // stride hi
  g1[7] = 0;
}
__device__ inline void tdm_load(const v4u& g0, const v8i_t& g1){
  v4i_t z4 = {0,0,0,0};
#if defined(__clang_major__) && (__clang_major__ >= 23)
  v8i_t z8 = {0,0,0,0,0,0,0,0};
  __builtin_amdgcn_tensor_load_to_lds(g0, g1, z4, z4, z8, 0);
#else
  __builtin_amdgcn_tensor_load_to_lds(g0, g1, z4, z4, 0);
#endif
}

// ---------------- kernels ----------------
// ws u32 slots: [0,1]=q min/max [2,3]=k [4,5]=v [6,7]=qk [8,9]=attn [10,11]=av
// ws f32 slots (resolved min/max pairs): 16..27
__global__ __launch_bounds__(32) void init_ws_kernel(unsigned* ws){
  int t = threadIdx.x;
  if (t < 12) ws[t] = (t & 1) ? 0x007FFFFFu /*enc(-inf)*/ : 0xFF800000u /*enc(+inf)*/;
}

__global__ __launch_bounds__(256) void minmax_kernel(const float4* __restrict__ x,
                                                     int n4, unsigned* __restrict__ slots){
  int tid = threadIdx.x, lane = tid & 31, wid = tid >> 5;
  float mn = INFINITY, mx = -INFINITY;
  for (int i = blockIdx.x * blockDim.x + tid; i < n4; i += gridDim.x * blockDim.x){
    float4 v = x[i];                              // global_load_b128
    mn = fminf(mn, fminf(fminf(v.x, v.y), fminf(v.z, v.w)));
    mx = fmaxf(mx, fmaxf(fmaxf(v.x, v.y), fmaxf(v.z, v.w)));
  }
  mn = wave_min(mn); mx = wave_max(mx);
  __shared__ float smn[8], smx[8];
  if (lane == 0){ smn[wid] = mn; smx[wid] = mx; }
  __syncthreads();
  if (tid == 0){
    for (int w = 1; w < 8; ++w){ mn = fminf(mn, smn[w]); mx = fmaxf(mx, smx[w]); }
    atomicMin(&slots[0], fenc(mn));
    atomicMax(&slots[1], fenc(mx));
  }
}

// EMA / fresh-stat resolution:  out = isfinite(old) ? 0.9*old + 0.1*cur : cur
__global__ void resolve_kernel(const unsigned* __restrict__ enc,
                               const float* __restrict__ oldmm,
                               float* __restrict__ outmm){
  if (threadIdx.x == 0 && blockIdx.x == 0){
    float c0 = fdec(enc[0]), c1 = fdec(enc[1]);
    float o0 = oldmm[0], o1 = oldmm[1];
    bool fin = __builtin_isfinite(o0) && __builtin_isfinite(o1);
    outmm[0] = fin ? (0.9f*o0 + 0.1f*c0) : c0;
    outmm[1] = fin ? (0.9f*o1 + 0.1f*c1) : c1;
  }
}

// Fused QK^T: raw product (bf16 WMMA) -> qk min/max stats; quantized product
// (fq(Q) x fq_lut(K^T), bf16 WMMA) -> attnW (pre-round, pre-scale).
// Double-buffered TDM pipeline over 4 k-chunks of 32; staged f32 chunks are
// converted once into fragment-native bf16 tiles (raw + quant separate).
__global__ __launch_bounds__(128) void qk_kernel(const float* __restrict__ Q,
    const float* __restrict__ K, const float* __restrict__ qmm,
    const float* __restrict__ kmm, float* __restrict__ attnW,
    unsigned* __restrict__ qkSlots){
  __shared__ __align__(16) float    stageQ[2][64*32];   // 2x8 KB
  __shared__ __align__(16) float    stageK[2][64*32];   // 2x8 KB
  __shared__ __align__(16) unsigned bQraw[64*16];       // 64x32 bf16 tiles, 4 KB each
  __shared__ __align__(16) unsigned bQq[64*16];
  __shared__ __align__(16) unsigned bKraw[64*16];
  __shared__ __align__(16) unsigned bKq[64*16];
  const int h = blockIdx.z, kvh = h >> 2;
  const int m0 = blockIdx.y * 64, n0 = blockIdx.x * 64;
  const int tid = threadIdx.x, wave = tid >> 5, lane = tid & 31;
  const int lm = lane & 15, hi16 = lane >> 4;
  const QParams qp = qload(qmm), kp = qload(kmm);
  const float kthr = QLUTMIN_ * fmaxf(fabsf(kp.lo), fabsf(kp.hi));

  if (wave == 0){                      // prologue: chunk 0 into buffer 0
    v4u g0; v8i_t g1;
    tdm_desc(g0, g1, lds_off(stageQ[0]), Q + (size_t)(h*SEQ + m0)*DH, 32, 64, DH, DH);
    tdm_load(g0, g1);
    tdm_desc(g0, g1, lds_off(stageK[0]), K + (size_t)(kvh*SEQ + n0)*DH, 32, 64, DH, DH);
    tdm_load(g0, g1);
  }

  v8f z8 = {0.f,0.f,0.f,0.f,0.f,0.f,0.f,0.f};
  v8f craw[4] = {z8,z8,z8,z8};
  v8f cq[4]   = {z8,z8,z8,z8};

  for (int kc = 0; kc < 4; ++kc){      // K-depth: DH=128 in 4x bf16 k=32 steps
    const int buf = kc & 1;
    if (wave == 0){
      if (kc + 1 < 4){                 // issue next chunk into other buffer
        const int kb = (kc + 1) * 32;
        v4u g0; v8i_t g1;
        tdm_desc(g0, g1, lds_off(stageQ[buf ^ 1]),
                 Q + (size_t)(h*SEQ + m0)*DH + kb, 32, 64, DH, DH);
        tdm_load(g0, g1);
        tdm_desc(g0, g1, lds_off(stageK[buf ^ 1]),
                 K + (size_t)(kvh*SEQ + n0)*DH + kb, 32, 64, DH, DH);
        tdm_load(g0, g1);
        __builtin_amdgcn_s_wait_tensorcnt(2);   // chunk kc's 2 ops done (in order)
      } else {
        __builtin_amdgcn_s_wait_tensorcnt(0);
      }
    }
    __syncthreads();

    // conversion: f32 quad (b128 load) -> raw/quant bf16 pairs (b64 stores)
#pragma unroll
    for (int t = tid; t < 512; t += 128){
      const int i4 = t * 4;
      float4 q = *(const float4*)&stageQ[buf][i4];
      uint2 qr = { pk2(q.x, q.y), pk2(q.z, q.w) };
      uint2 qq = { pk2(fq_u(q.x, qp), fq_u(q.y, qp)),
                   pk2(fq_u(q.z, qp), fq_u(q.w, qp)) };
      *(uint2*)&bQraw[2*t] = qr;
      *(uint2*)&bQq[2*t]   = qq;
      float4 k = *(const float4*)&stageK[buf][i4];
      uint2 kr = { pk2(k.x, k.y), pk2(k.z, k.w) };
      uint2 kq = { pk2(fq_lut(k.x, kp, kthr), fq_lut(k.y, kp, kthr)),
                   pk2(fq_lut(k.z, kp, kthr), fq_lut(k.w, kp, kthr)) };
      *(uint2*)&bKraw[2*t] = kr;
      *(uint2*)&bKq[2*t]   = kq;
    }
    __syncthreads();

    // fragment loads: contiguous b128s straight into WMMA operands
    const int arow = (wave*16 + lm) * 16 + hi16*4;
    v16bf araw = frag44(&bQraw[arow], &bQraw[arow + 8]);
    v16bf aq   = frag44(&bQq[arow],   &bQq[arow + 8]);
#pragma unroll
    for (int nt = 0; nt < 4; ++nt){
      const int boff = (nt*16 + lm) * 16 + hi16*4;
      v16bf braw = frag44(&bKraw[boff], &bKraw[boff + 8]);
      v16bf bq   = frag44(&bKq[boff],   &bKq[boff + 8]);
      craw[nt] = __builtin_amdgcn_wmma_f32_16x16x32_bf16(false, araw, false, braw,
                                                         (short)0, craw[nt], false, false);
      cq[nt]   = __builtin_amdgcn_wmma_f32_16x16x32_bf16(false, aq, false, bq,
                                                         (short)0, cq[nt], false, false);
    }
    __syncthreads();                   // tiles/stage reusable next iteration
  }

  float rmn = INFINITY, rmx = -INFINITY;
#pragma unroll
  for (int nt = 0; nt < 4; ++nt){
#pragma unroll
    for (int r = 0; r < 8; ++r){       // C/D layout: row = r + 8*hi16, col = lm
      int gm = m0 + wave*16 + r + 8*hi16;
      int gn = n0 + nt*16 + lm;
      float raw = craw[nt][r] * SCALING_;
      rmn = fminf(rmn, raw); rmx = fmaxf(rmx, raw);
      attnW[((size_t)(h*SEQ) + gm)*SEQ + gn] = cq[nt][r];
    }
  }
  rmn = wave_min(rmn); rmx = wave_max(rmx);
  if (lane == 0){
    atomicMin(&qkSlots[0], fenc(rmn));
    atomicMax(&qkSlots[1], fenc(rmx));
  }
}

// Per-row: stochastic-round quant product with qk_mm, *scale, +mask, softmax,
// attn min/max stats. One block per (h, i) row.
__global__ __launch_bounds__(256) void softmax_kernel(float* __restrict__ attnW,
    const float* __restrict__ mask, const float* __restrict__ qkmm,
    unsigned* __restrict__ aSlots){
  const int row = blockIdx.x;          // h*SEQ + i
  const int i = row & (SEQ - 1);
  const int tid = threadIdx.x, lane = tid & 31, wid = tid >> 5;
  const QParams p = qload(qkmm);
  const size_t rbase = (size_t)row * SEQ;

  float v[8];
#pragma unroll
  for (int c = 0; c < 8; ++c){
    int j = c*256 + tid;
    float x = attnW[rbase + j];
    float t = (fminf(fmaxf(x, p.lo), p.hi) - p.lo) * p.inv;
    float u = urand01((unsigned)(rbase + j), 0x9E3779B9u);
    float q = floorf(t + u) * p.scale + p.lo;
    v[c] = q * SCALING_ + mask[(size_t)i*SEQ + j];
  }
  __shared__ float red[8];
  float m = v[0];
#pragma unroll
  for (int c = 1; c < 8; ++c) m = fmaxf(m, v[c]);
  m = wave_max(m);
  if (lane == 0) red[wid] = m;
  __syncthreads();
  float bm = red[0];
  for (int w = 1; w < 8; ++w) bm = fmaxf(bm, red[w]);
  __syncthreads();
  float s = 0.f;
#pragma unroll
  for (int c = 0; c < 8; ++c) s += __expf(v[c] - bm);
  s = wave_sum(s);
  if (lane == 0) red[wid] = s;
  __syncthreads();
  float bs = 0.f;
  for (int w = 0; w < 8; ++w) bs += red[w];
  float inv = 1.0f / bs;

  float pmn = INFINITY, pmx = -INFINITY;
#pragma unroll
  for (int c = 0; c < 8; ++c){
    int j = c*256 + tid;
    float pr = __expf(v[c] - bm) * inv;
    attnW[rbase + j] = pr;
    pmn = fminf(pmn, pr); pmx = fmaxf(pmx, pr);
  }
  pmn = wave_min(pmn); pmx = wave_max(pmx);
  __syncthreads();
  __shared__ float red2[8];
  if (lane == 0){ red[wid] = pmn; red2[wid] = pmx; }
  __syncthreads();
  if (tid == 0){
    for (int w = 1; w < 8; ++w){ pmn = fminf(pmn, red[w]); pmx = fmaxf(pmx, red2[w]); }
    atomicMin(&aSlots[0], fenc(pmn));
    atomicMax(&aSlots[1], fenc(pmx));
  }
}

// Fused AV: raw attn@V -> av stats; quantized fq(attn)@fq_lut(V) -> out0
// ([S,H,D], pre-round). Double-buffered TDM pipeline over 64 K-chunks of 32;
// V chunk is transposed during conversion so B fragments load contiguously.
__global__ __launch_bounds__(128) void av_kernel(const float* __restrict__ attnW,
    const float* __restrict__ V, const float* __restrict__ amm,
    const float* __restrict__ vmm, float* __restrict__ out0,
    unsigned* __restrict__ avSlots){
  __shared__ __align__(16) float    stageA[2][64*32];   // 2x8 KB
  __shared__ __align__(16) float    stageV[2][32*DH];   // 2x16 KB
  __shared__ __align__(16) unsigned bAraw[64*16];       // 64x32 bf16, 4 KB each
  __shared__ __align__(16) unsigned bAq[64*16];
  __shared__ __align__(16) unsigned bVtraw[128*16];     // Vt: 128x32 bf16, 8 KB each
  __shared__ __align__(16) unsigned bVtq[128*16];
  const int m0 = blockIdx.x * 64;
  const int h = blockIdx.y, kvh = h >> 2;
  const int tid = threadIdx.x, wave = tid >> 5, lane = tid & 31;
  const int lm = lane & 15, hi16 = lane >> 4;
  const QParams ap = qload(amm), vp = qload(vmm);
  const float vthr = QLUTMIN_ * fmaxf(fabsf(vp.lo), fabsf(vp.hi));

  v8f z8 = {0.f,0.f,0.f,0.f,0.f,0.f,0.f,0.f};
  v8f craw[8] = {z8,z8,z8,z8,z8,z8,z8,z8};
  v8f cq[8]   = {z8,z8,z8,z8,z8,z8,z8,z8};

  if (wave == 0){                      // prologue: chunk 0 into buffer 0
    v4u g0; v8i_t g1;
    tdm_desc(g0, g1, lds_off(stageA[0]),
             attnW + ((size_t)(h*SEQ + m0))*SEQ, 32, 64, SEQ, SEQ);
    tdm_load(g0, g1);
    tdm_desc(g0, g1, lds_off(stageV[0]),
             V + (size_t)(kvh*SEQ)*DH, 32*DH, 0, 32*DH, 32*DH);
    tdm_load(g0, g1);
  }

  for (int kc = 0; kc < 64; ++kc){
    const int buf = kc & 1;
    if (wave == 0){
      if (kc + 1 < 64){                // issue next chunk into other buffer
        const int k0n = (kc + 1) * 32;
        v4u g0; v8i_t g1;
        tdm_desc(g0, g1, lds_off(stageA[buf ^ 1]),
                 attnW + ((size_t)(h*SEQ + m0))*SEQ + k0n, 32, 64, SEQ, SEQ);
        tdm_load(g0, g1);
        tdm_desc(g0, g1, lds_off(stageV[buf ^ 1]),
                 V + (size_t)(kvh*SEQ + k0n)*DH, 32*DH, 0, 32*DH, 32*DH);
        tdm_load(g0, g1);
        __builtin_amdgcn_s_wait_tensorcnt(2);   // chunk kc's 2 ops done (in order)
      } else {
        __builtin_amdgcn_s_wait_tensorcnt(0);
      }
    }
    __syncthreads();

    // attn chunk: f32 quads -> raw/quant bf16 pairs (row-major, b64 stores)
#pragma unroll
    for (int t = tid; t < 512; t += 128){
      const int i4 = t * 4;
      float4 a = *(const float4*)&stageA[buf][i4];
      uint2 ar = { pk2(a.x, a.y), pk2(a.z, a.w) };
      uint2 aq2 = { pk2(fq_u(a.x, ap), fq_u(a.y, ap)),
                    pk2(fq_u(a.z, ap), fq_u(a.w, ap)) };
      *(uint2*)&bAraw[2*t] = ar;
      *(uint2*)&bAq[2*t]   = aq2;
    }
    // V chunk: transpose to Vt[n][k] with k-pairs so one b32 store per pair
#pragma unroll
    for (int t = tid; t < 2048; t += 128){
      const int n = t >> 4, kp = (t & 15) * 2;
      float v0 = stageV[buf][kp*DH + n];
      float v1 = stageV[buf][(kp+1)*DH + n];
      bVtraw[n*16 + (kp>>1)] = pk2(v0, v1);
      bVtq[n*16 + (kp>>1)]   = pk2(fq_lut(v0, vp, vthr), fq_lut(v1, vp, vthr));
    }
    __syncthreads();

    const int arow = (wave*16 + lm) * 16 + hi16*4;
    v16bf araw = frag44(&bAraw[arow], &bAraw[arow + 8]);
    v16bf aq   = frag44(&bAq[arow],   &bAq[arow + 8]);
#pragma unroll
    for (int nt = 0; nt < 8; ++nt){
      const int boff = (nt*16 + lm) * 16 + hi16*4;
      v16bf braw = frag44(&bVtraw[boff], &bVtraw[boff + 8]);
      v16bf bq   = frag44(&bVtq[boff],   &bVtq[boff + 8]);
      craw[nt] = __builtin_amdgcn_wmma_f32_16x16x32_bf16(false, araw, false, braw,
                                                         (short)0, craw[nt], false, false);
      cq[nt]   = __builtin_amdgcn_wmma_f32_16x16x32_bf16(false, aq, false, bq,
                                                         (short)0, cq[nt], false, false);
    }
    __syncthreads();                   // buffers reusable next iteration
  }

  float rmn = INFINITY, rmx = -INFINITY;
#pragma unroll
  for (int nt = 0; nt < 8; ++nt){
#pragma unroll
    for (int r = 0; r < 8; ++r){
      int gm = m0 + wave*16 + r + 8*hi16;
      int gd = nt*16 + lm;
      float raw = craw[nt][r];
      rmn = fminf(rmn, raw); rmx = fmaxf(rmx, raw);
      out0[((size_t)gm*NH + h)*DH + gd] = cq[nt][r];   // [B,S,H,D]
    }
  }
  rmn = wave_min(rmn); rmx = wave_max(rmx);
  if (lane == 0){
    atomicMin(&avSlots[0], fenc(rmn));
    atomicMax(&avSlots[1], fenc(rmx));
  }
}

// elementwise stochastic rounding of attn_output with av_mm (in place, float4)
__global__ __launch_bounds__(256) void finalize_kernel(float4* __restrict__ out0,
    const float* __restrict__ avmm, int n4){
  const QParams p = qload(avmm);
  for (int i = blockIdx.x * blockDim.x + threadIdx.x; i < n4; i += gridDim.x * blockDim.x){
    float4 x = out0[i];
    float* xp = &x.x;
#pragma unroll
    for (int c = 0; c < 4; ++c){
      float t = (fminf(fmaxf(xp[c], p.lo), p.hi) - p.lo) * p.inv;
      float u = urand01((unsigned)(i*4 + c), 0x85EBCA6Bu);
      xp[c] = floorf(t + u) * p.scale + p.lo;
    }
    out0[i] = x;
  }
}

// ---------------- host launcher ----------------
extern "C" void kernel_launch(void* const* d_in, const int* in_sizes, int n_in,
                              void* d_out, int out_size, void* d_ws, size_t ws_size,
                              hipStream_t stream){
  (void)in_sizes; (void)n_in; (void)out_size; (void)ws_size;
  const float* query  = (const float*)d_in[0];
  const float* key    = (const float*)d_in[1];
  const float* value  = (const float*)d_in[2];
  const float* mask   = (const float*)d_in[3];
  const float* q_old  = (const float*)d_in[4];
  const float* k_old  = (const float*)d_in[5];
  const float* qk_old = (const float*)d_in[6];
  const float* a_old  = (const float*)d_in[7];
  const float* v_old  = (const float*)d_in[8];
  const float* av_old = (const float*)d_in[9];

  float*    out0  = (float*)d_out;          // attn_output [1,2048,32,128]
  float*    attnW = out0 + (size_t)OUT0_N;  // attn_weights [1,32,2048,2048]
  unsigned* wsu   = (unsigned*)d_ws;
  float*    wsf   = (float*)d_ws;

  init_ws_kernel<<<1, 32, 0, stream>>>(wsu);
  minmax_kernel<<<1024, 256, 0, stream>>>((const float4*)query, (NH*SEQ*DH)/4,  wsu + 0);
  minmax_kernel<<<256,  256, 0, stream>>>((const float4*)key,   (NKV*SEQ*DH)/4, wsu + 2);
  minmax_kernel<<<256,  256, 0, stream>>>((const float4*)value, (NKV*SEQ*DH)/4, wsu + 4);
  resolve_kernel<<<1, 1, 0, stream>>>(wsu + 0, q_old, wsf + 16);
  resolve_kernel<<<1, 1, 0, stream>>>(wsu + 2, k_old, wsf + 18);
  resolve_kernel<<<1, 1, 0, stream>>>(wsu + 4, v_old, wsf + 20);

  qk_kernel<<<dim3(SEQ/64, SEQ/64, NH), 128, 0, stream>>>(
      query, key, wsf + 16, wsf + 18, attnW, wsu + 6);
  resolve_kernel<<<1, 1, 0, stream>>>(wsu + 6, qk_old, wsf + 22);

  softmax_kernel<<<NH*SEQ, 256, 0, stream>>>(attnW, mask, wsf + 22, wsu + 8);
  resolve_kernel<<<1, 1, 0, stream>>>(wsu + 8, a_old, wsf + 24);

  av_kernel<<<dim3(SEQ/64, NH), 128, 0, stream>>>(
      attnW, value, wsf + 24, wsf + 20, out0, wsu + 10);
  resolve_kernel<<<1, 1, 0, stream>>>(wsu + 10, av_old, wsf + 26);

  finalize_kernel<<<2048, 256, 0, stream>>>((float4*)out0, wsf + 26, OUT0_N/4);
}
